// Model_41781441856030
// MI455X (gfx1250) — matrix operand order repaired and checked by reading, not compiled, not run on hardware
//
#include <hip/hip_runtime.h>
#include <hip/hip_bf16.h>
#include <stdint.h>

// Problem dims
#define BB 64
#define TT 256
#define HD 768
#define HHD 384
#define G4 1536   // 4*HH
#define NK 9

typedef __attribute__((ext_vector_type(16))) _Float16 v16h;
typedef __attribute__((ext_vector_type(8)))  _Float16 v8h;
typedef __attribute__((ext_vector_type(4)))  _Float16 v4h;
typedef __attribute__((ext_vector_type(8)))  float    v8f;

union frag16 { v16h v; v8h h[2]; };

// A-matrix fragment (16x32 f16, MxK), source row-major [M, stride] at tile base.
// Per ISA: lane m=lane&15; half=lane>>4; elems 0..7 = K(half*8 .. +7), 8..15 = K(16+half*8 .. +7)
__device__ __forceinline__ v16h ld_frag_a(const _Float16* base, int stride) {
  const int lane = threadIdx.x & 31;
  const _Float16* p = base + (lane & 15) * stride + ((lane >> 4) << 3);
  frag16 u;
  u.h[0] = *(const v8h*)(p);
  u.h[1] = *(const v8h*)(p + 16);
  return u.v;
}

// B-matrix fragment (32x16 f16, KxN), source is N-major rows [N, stride] (i.e. W itself).
// lane n=lane&15; half=lane>>4; elems 0..7 = K(half*8 .. +7), 8..15 = K(16+half*8 .. +7)
__device__ __forceinline__ v16h ld_frag_b(const _Float16* base, int stride) {
  const int lane = threadIdx.x & 31;
  const _Float16* p = base + (lane & 15) * stride + ((lane >> 4) << 3);
  frag16 u;
  u.h[0] = *(const v8h*)(p);
  u.h[1] = *(const v8h*)(p + 16);
  return u.v;
}

#define WMMA16(a, b, c) \
  __builtin_amdgcn_wmma_f32_16x16x32_f16(false, (a), false, (b), (short)0, (c), false, false)

// CDNA5 has a hardware TANH transcendental (TRANS32 pipe, co-executes with XDL).
__device__ __forceinline__ float fast_tanh(float x) {
#if __has_builtin(__builtin_amdgcn_tanhf)
  return __builtin_amdgcn_tanhf(x);
#elif __has_builtin(__builtin_amdgcn_tanh_f32)
  return __builtin_amdgcn_tanh_f32(x);
#else
  const float e = __expf(2.0f * x);
  return (e - 1.0f) / (e + 1.0f);
#endif
}

__device__ __forceinline__ float sigf(float x) {
#if __has_builtin(__builtin_amdgcn_rcpf)
  return __builtin_amdgcn_rcpf(1.0f + __expf(-x));
#else
  return 1.0f / (1.0f + __expf(-x));
#endif
}

// ---------------------------------------------------------------- converts
__global__ void k_cvt_f16(_Float16* __restrict__ dst, const float* __restrict__ src, int n) {
  int i = blockIdx.x * blockDim.x + threadIdx.x;
  if (i < n) dst[i] = (_Float16)src[i];
}

__global__ void k_bias_sum(float* __restrict__ dst, const float* __restrict__ a,
                           const float* __restrict__ b, int n) {
  int i = blockIdx.x * blockDim.x + threadIdx.x;
  if (i < n) dst[i] = a[i] + b[i];
}

// ---------------------------------------------------------------- batched input projection
// A: f16 [16384, K] row-major (row r = b*T + t); W: f16 [1536, K] (N-major); bias f32 [1536]
// out: f16 gate-interleaved [T, B, 384, 4]:
//   out[(((r&255)*64 + (r>>8))*384 + n%384)*4 + n/384] = (A.W^T)[r][n] + bias[n]
// A 16-wide n-tile never straddles a 384 (gate) boundary, so gate is tile-uniform.
#define GT_LDS 40  // 32 + 8 pad halves
__launch_bounds__(256)
__global__ void k_gemm_xproj(const _Float16* __restrict__ A, const _Float16* __restrict__ W,
                             const float* __restrict__ bias, _Float16* __restrict__ out,
                             int Kdim) {
  __shared__ _Float16 As[128 * GT_LDS];
  __shared__ _Float16 Bs[128 * GT_LDS];
  const int tid = threadIdx.x;
  const int wv  = tid >> 5;
  const int m0  = blockIdx.y * 128;
  const int n0  = blockIdx.x * 128;
  const int wm  = (wv >> 1) * 2;   // local m-tile base: 2 tiles
  const int wn  = (wv & 1) * 4;    // local n-tile base: 4 tiles
  v8f acc[2][4] = {};
  const int lr = tid >> 1;         // 0..127
  const int lc = (tid & 1) * 16;   // 0 / 16

  for (int k0 = 0; k0 < Kdim; k0 += 32) {
    const _Float16* ap = &A[(size_t)(m0 + lr) * Kdim + k0 + lc];
    const _Float16* bp = &W[(size_t)(n0 + lr) * Kdim + k0 + lc];
    *(v8h*)&As[lr * GT_LDS + lc]     = *(const v8h*)(ap);
    *(v8h*)&As[lr * GT_LDS + lc + 8] = *(const v8h*)(ap + 8);
    *(v8h*)&Bs[lr * GT_LDS + lc]     = *(const v8h*)(bp);
    *(v8h*)&Bs[lr * GT_LDS + lc + 8] = *(const v8h*)(bp + 8);
    __syncthreads();
    v16h af0 = ld_frag_a(&As[(wm * 16) * GT_LDS], GT_LDS);
    v16h af1 = ld_frag_a(&As[(wm * 16 + 16) * GT_LDS], GT_LDS);
#pragma unroll
    for (int nt = 0; nt < 4; ++nt) {
      v16h bf = ld_frag_b(&Bs[((wn + nt) * 16) * GT_LDS], GT_LDS);
      acc[0][nt] = WMMA16(af0, bf, acc[0][nt]);
      acc[1][nt] = WMMA16(af1, bf, acc[1][nt]);
    }
    __syncthreads();
  }
  const int lane  = tid & 31;
  const int cn    = lane & 15;
  const int rbase = (lane >> 4) * 8;
#pragma unroll
  for (int mt = 0; mt < 2; ++mt)
#pragma unroll
    for (int nt = 0; nt < 4; ++nt) {
      const int n    = n0 + (wn + nt) * 16 + cn;
      const int gate = n / HHD;
      const int col  = n - gate * HHD;
      const float bv = bias[n];
#pragma unroll
      for (int i = 0; i < 8; ++i) {
        const int r    = m0 + (wm + mt) * 16 + rbase + i;
        const int orow = (r & (TT - 1)) * BB + (r >> 8);  // [t][b]
        out[((((size_t)orow * HHD) + col) << 2) + gate] = (_Float16)(acc[mt][nt][i] + bv);
      }
    }
}

// ---------------------------------------------------------------- persistent bi-LSTM recurrence
// gridDim.x = 2 (dir 0=fwd, 1=bwd), 256 threads = 8 waves.
// xproj: f16 [T, B, 384, 4] gate-interleaved (bias folded); whh: f16 [1536, 384] (N-major);
// Hout: f16 [B, T, 768], dir d writes columns [d*384, d*384+384).
//
// Each wave owns 48 h-columns (3 n-tiles) and computes all 4 gates for them,
// so the cell/h update is wave-local.  Loop nest keeps 4x4 (m-tile x gate)
// accumulators live so every W_hh fragment is loaded exactly once per step
// (1.18 MB/step from L2 = the minimum).
#define HSTR 392   // 384 + 8 pad halves
#define CSTR 400   // 384 + 16 pad floats
__launch_bounds__(256)
__global__ void k_lstm(const _Float16* __restrict__ xproj_f, const _Float16* __restrict__ xproj_b,
                       const _Float16* __restrict__ whh_f, const _Float16* __restrict__ whh_b,
                       _Float16* __restrict__ Hout) {
  extern __shared__ char smem[];
  _Float16* hbuf0 = (_Float16*)smem;
  _Float16* hbuf1 = hbuf0 + BB * HSTR;
  float*    cbuf  = (float*)(smem + (size_t)2 * BB * HSTR * sizeof(_Float16));

  const int dir = blockIdx.x;
  const _Float16* xproj = dir ? xproj_b : xproj_f;
  const _Float16* whh   = dir ? whh_b   : whh_f;

  for (int i = threadIdx.x; i < BB * HSTR; i += 256) hbuf0[i] = (_Float16)0.0f;
  for (int i = threadIdx.x; i < BB * CSTR; i += 256) cbuf[i] = 0.0f;
  __syncthreads();

  const int wv   = threadIdx.x >> 5;
  const int lane = threadIdx.x & 31;
  const int cn   = lane & 15;
  const int rb   = (lane >> 4) * 8;

  _Float16* hp = hbuf0;
  _Float16* hn = hbuf1;

  for (int step = 0; step < TT; ++step) {
    const int t = dir ? (TT - 1 - step) : step;

    // Prefetch next step's x-projection slab (196 KB, streamed from HBM/L2).
    const int tn = dir ? (t - 1) : (t + 1);
    if (tn >= 0 && tn < TT) {
      const char* pf = (const char*)(xproj + (size_t)tn * BB * G4);
#pragma unroll
      for (int q = 0; q < 3; ++q)
        __builtin_prefetch(pf + ((size_t)(threadIdx.x + q * 256)) * 256, 0, 1);
    }

    for (int j = 0; j < 3; ++j) {              // 3 n-tiles of 16 h-columns per wave
      const int col0 = (wv * 3 + j) * 16;
      v8f acc[4][4] = {};                      // [m-tile][gate i,f,g,o]
      for (int kt = 0; kt < HHD / 32; ++kt) {
        v16h af[4];
#pragma unroll
        for (int mt = 0; mt < 4; ++mt)
          af[mt] = ld_frag_a(hp + (mt * 16) * HSTR + kt * 32, HSTR);
#pragma unroll
        for (int g = 0; g < 4; ++g) {
          v16h bf = ld_frag_b(whh + (size_t)(g * HHD + col0) * HHD + kt * 32, HHD);
#pragma unroll
          for (int mt = 0; mt < 4; ++mt)
            acc[mt][g] = WMMA16(af[mt], bf, acc[mt][g]);
        }
      }
      const int col = col0 + cn;
#pragma unroll
      for (int mt = 0; mt < 4; ++mt) {
#pragma unroll
        for (int i = 0; i < 8; ++i) {
          const int row = mt * 16 + rb + i;    // batch index
          // one 8-byte load fetches all four gate pre-activations
          const v4h xg = *(const v4h*)&xproj[(((size_t)t * BB + row) * HHD + col) << 2];
          float gi = acc[mt][0][i] + (float)xg[0];
          float gf = acc[mt][1][i] + (float)xg[1];
          float gg = acc[mt][2][i] + (float)xg[2];
          float go = acc[mt][3][i] + (float)xg[3];
          float c = cbuf[row * CSTR + col];
          c = sigf(gf) * c + sigf(gi) * fast_tanh(gg);
          cbuf[row * CSTR + col] = c;
          float h = sigf(go) * fast_tanh(c);
          hn[row * HSTR + col] = (_Float16)h;
          Hout[((size_t)row * TT + t) * HD + dir * HHD + col] = (_Float16)h;
        }
      }
    }
    __syncthreads();
    _Float16* tmp = hp; hp = hn; hn = tmp;
  }
}

// ---------------------------------------------------------------- FC: em = H1 @ fc_w^T + fc_b
__launch_bounds__(256)
__global__ void k_fc(const _Float16* __restrict__ Hx, const float* __restrict__ fcw,
                     const float* __restrict__ fcb, float* __restrict__ em) {
  __shared__ float wsm[NK * HD];
  for (int i = threadIdx.x; i < NK * HD; i += 256) wsm[i] = fcw[i];
  __syncthreads();
  const int r = blockIdx.x * blockDim.x + threadIdx.x;
  if (r >= BB * TT) return;
  float acc[NK];
#pragma unroll
  for (int k = 0; k < NK; ++k) acc[k] = fcb[k];
  for (int j0 = 0; j0 < HD; j0 += 8) {
    const v8h x8 = *(const v8h*)&Hx[(size_t)r * HD + j0];
#pragma unroll
    for (int q = 0; q < 8; ++q) {
      const float x = (float)x8[q];
#pragma unroll
      for (int k = 0; k < NK; ++k) acc[k] += x * wsm[k * HD + j0 + q];
    }
  }
#pragma unroll
  for (int k = 0; k < NK; ++k) em[(size_t)r * NK + k] = acc[k];
}

// ---------------------------------------------------------------- CRF log-likelihood (1 block, 64 thr)
__global__ void k_crf(const float* __restrict__ em, const int* __restrict__ y,
                      const float* __restrict__ start, const float* __restrict__ trans,
                      const float* __restrict__ endt, float* __restrict__ out) {
  __shared__ float tr[NK * NK];
  __shared__ float red[BB];
  if (threadIdx.x < NK * NK) tr[threadIdx.x] = trans[threadIdx.x];
  __syncthreads();
  const int b = threadIdx.x;
  const int yb = b * TT;
  // numerator
  float num = start[y[yb]] + endt[y[yb + TT - 1]];
  for (int t = 0; t < TT; ++t)     num += em[((size_t)b * TT + t) * NK + y[yb + t]];
  for (int t = 0; t < TT - 1; ++t) num += tr[y[yb + t] * NK + y[yb + t + 1]];
  // denominator: forward logsumexp
  float a[NK];
#pragma unroll
  for (int j = 0; j < NK; ++j) a[j] = start[j] + em[(size_t)b * TT * NK + j];
  for (int t = 1; t < TT; ++t) {
    float an[NK];
#pragma unroll
    for (int j = 0; j < NK; ++j) {
      float m = -1e30f;
#pragma unroll
      for (int i = 0; i < NK; ++i) m = fmaxf(m, a[i] + tr[i * NK + j]);
      float s = 0.0f;
#pragma unroll
      for (int i = 0; i < NK; ++i) s += __expf(a[i] + tr[i * NK + j] - m);
      an[j] = m + __logf(s) + em[((size_t)b * TT + t) * NK + j];
    }
#pragma unroll
    for (int j = 0; j < NK; ++j) a[j] = an[j];
  }
  float m = -1e30f;
#pragma unroll
  for (int j = 0; j < NK; ++j) m = fmaxf(m, a[j] + endt[j]);
  float s = 0.0f;
#pragma unroll
  for (int j = 0; j < NK; ++j) s += __expf(a[j] + endt[j] - m);
  red[b] = num - (m + __logf(s));
  __syncthreads();
  if (b == 0) {
    float tot = 0.0f;
    for (int i = 0; i < BB; ++i) tot += red[i];
    out[0] = tot;
  }
}

// ---------------------------------------------------------------- Viterbi (1 block, 64 thr)
__global__ void k_viterbi(const float* __restrict__ em, const float* __restrict__ start,
                          const float* __restrict__ trans, const float* __restrict__ endt,
                          signed char* __restrict__ hist, float* __restrict__ out_tags) {
  __shared__ float tr[NK * NK];
  if (threadIdx.x < NK * NK) tr[threadIdx.x] = trans[threadIdx.x];
  __syncthreads();
  const int b = threadIdx.x;
  float s[NK];
#pragma unroll
  for (int j = 0; j < NK; ++j) s[j] = start[j] + em[(size_t)b * TT * NK + j];
  for (int t = 1; t < TT; ++t) {
    float sn[NK];
#pragma unroll
    for (int j = 0; j < NK; ++j) {
      float best = -1e30f; int bi = 0;
#pragma unroll
      for (int i = 0; i < NK; ++i) {
        float v = s[i] + tr[i * NK + j];
        if (v > best) { best = v; bi = i; }
      }
      sn[j] = best + em[((size_t)b * TT + t) * NK + j];
      hist[((size_t)b * TT + t) * NK + j] = (signed char)bi;
    }
#pragma unroll
    for (int j = 0; j < NK; ++j) s[j] = sn[j];
  }
  float best = -1e30f; int cur = 0;
#pragma unroll
  for (int j = 0; j < NK; ++j) {
    float v = s[j] + endt[j];
    if (v > best) { best = v; cur = j; }
  }
  out_tags[b * TT + TT - 1] = (float)cur;
  for (int t = TT - 1; t >= 1; --t) {
    cur = hist[((size_t)b * TT + t) * NK + cur];
    out_tags[b * TT + t - 1] = (float)cur;
  }
}

// ---------------------------------------------------------------- host launcher
extern "C" void kernel_launch(void* const* d_in, const int* in_sizes, int n_in,
                              void* d_out, int out_size, void* d_ws, size_t ws_size,
                              hipStream_t stream) {
  const float* bert = (const float*)d_in[0];
  const float* wih[2][2]; const float* whh[2][2];
  const float* bih[2][2]; const float* bhh[2][2];
  for (int l = 0; l < 2; ++l)
    for (int d = 0; d < 2; ++d) {
      const int base = 1 + l * 8 + d * 4;
      wih[l][d] = (const float*)d_in[base + 0];
      whh[l][d] = (const float*)d_in[base + 1];
      bih[l][d] = (const float*)d_in[base + 2];
      bhh[l][d] = (const float*)d_in[base + 3];
    }
  const float* fcw    = (const float*)d_in[17];
  const float* fcb    = (const float*)d_in[18];
  const float* startt = (const float*)d_in[19];
  const float* trans  = (const float*)d_in[20];
  const float* endt   = (const float*)d_in[21];
  const int*   y      = (const int*)d_in[22];

  char* ws = (char*)d_ws;
  size_t off = 0;
  auto alloc = [&](size_t bytes) -> char* {
    char* p = ws + off;
    off = (off + bytes + 255) & ~(size_t)255;
    return p;
  };
  // Workspace layout (~166 MB total)
  _Float16* X16 = (_Float16*)alloc((size_t)BB * TT * HD * 2);        // also reused as H1
  _Float16* Wih16[2][2]; _Float16* Whh16[2][2]; float* bias[2][2];
  for (int l = 0; l < 2; ++l)
    for (int d = 0; d < 2; ++d) {
      Wih16[l][d] = (_Float16*)alloc((size_t)G4 * HD * 2);
      Whh16[l][d] = (_Float16*)alloc((size_t)G4 * HHD * 2);
      bias[l][d]  = (float*)alloc((size_t)G4 * 4);
    }
  _Float16* XPf = (_Float16*)alloc((size_t)BB * TT * G4 * 2);
  _Float16* XPb = (_Float16*)alloc((size_t)BB * TT * G4 * 2);
  _Float16* H0  = (_Float16*)alloc((size_t)BB * TT * HD * 2);
  float*    em  = (float*)alloc((size_t)BB * TT * NK * 4);
  signed char* hist = (signed char*)alloc((size_t)BB * TT * NK);
  _Float16* H1 = X16;  // alias: X16 dead after layer-0 projection

  // f32 -> f16 conversions + bias sums
  {
    int n = BB * TT * HD;
    k_cvt_f16<<<(n + 255) / 256, 256, 0, stream>>>(X16, bert, n);
    for (int l = 0; l < 2; ++l)
      for (int d = 0; d < 2; ++d) {
        int n1 = G4 * HD, n2 = G4 * HHD;
        k_cvt_f16<<<(n1 + 255) / 256, 256, 0, stream>>>(Wih16[l][d], wih[l][d], n1);
        k_cvt_f16<<<(n2 + 255) / 256, 256, 0, stream>>>(Whh16[l][d], whh[l][d], n2);
        k_bias_sum<<<(G4 + 255) / 256, 256, 0, stream>>>(bias[l][d], bih[l][d], bhh[l][d], G4);
      }
  }

  const dim3 ggrid(G4 / 128, (BB * TT) / 128);  // (12, 128)
  const size_t lstm_lds = (size_t)2 * BB * HSTR * sizeof(_Float16) + (size_t)BB * CSTR * sizeof(float);

  // Layer 0
  k_gemm_xproj<<<ggrid, 256, 0, stream>>>(X16, Wih16[0][0], bias[0][0], XPf, HD);
  k_gemm_xproj<<<ggrid, 256, 0, stream>>>(X16, Wih16[0][1], bias[0][1], XPb, HD);
  k_lstm<<<2, 256, lstm_lds, stream>>>(XPf, XPb, Whh16[0][0], Whh16[0][1], H0);
  // Layer 1
  k_gemm_xproj<<<ggrid, 256, 0, stream>>>(H0, Wih16[1][0], bias[1][0], XPf, HD);
  k_gemm_xproj<<<ggrid, 256, 0, stream>>>(H0, Wih16[1][1], bias[1][1], XPb, HD);
  k_lstm<<<2, 256, lstm_lds, stream>>>(XPf, XPb, Whh16[1][0], Whh16[1][1], H1);

  // Emissions + CRF + Viterbi
  k_fc<<<(BB * TT + 255) / 256, 256, 0, stream>>>(H1, fcw, fcb, em);
  k_crf<<<1, BB, 0, stream>>>(em, y, startt, trans, endt, (float*)d_out);
  k_viterbi<<<1, BB, 0, stream>>>(em, startt, trans, endt, hist, (float*)d_out + 1);
}
